// MetaBatchRelationalModule_48137993453732
// MI455X (gfx1250) — matrix operand order repaired and checked by reading, not compiled, not run on hardware
//
#include <hip/hip_runtime.h>

// CDNA5 / gfx1250 WMMA types
typedef __bf16 v16bf __attribute__((ext_vector_type(16)));
typedef float  v8f   __attribute__((ext_vector_type(8)));

#define B_SZ 256
#define C_SZ 256
#define L_SZ 25          // 5*5 locations
#define D_SZ 257         // C + coord
#define HID  64
#define OUTU 32
#define XS_ROWS 32       // 25 real rows + 7 zero pad rows (branch-free A gather)
#define XS_STRIDE 264    // bf16 elements per row (even -> 4B-aligned pairs)
#define UV_ROWS 32       // 25 real + 7 scratch rows (branch-free C store)
#define UV_STRIDE 66     // even -> 8B-aligned float2 rows; 66%64=2 spreads banks

__device__ __forceinline__ float leaky(float x) { return x >= 0.0f ? x : 0.01f * x; }

// A-matrix (16x32 bf16) K index for VGPR v, lane-half h (ISA 7.12.2). Always even.
__device__ __forceinline__ int a_kidx(int v, int h) {
    return (v < 4) ? (8 * h + 2 * v) : (16 + 8 * h + 2 * (v - 4));
}

union AVec { v16bf v; unsigned int u[8]; };

__global__ __launch_bounds__(128)
void relnet_kernel(const float* __restrict__ x_img,
                   const float* __restrict__ g0_w, const float* __restrict__ g0_b,
                   const float* __restrict__ g1_w, const float* __restrict__ g1_b,
                   const float* __restrict__ post_w, const float* __restrict__ post_b,
                   const float* __restrict__ out_w, const float* __restrict__ out_b,
                   float* __restrict__ out) {
    __shared__ __align__(16) unsigned short xs[XS_ROWS * XS_STRIDE]; // bf16 bits, (l, c)
    __shared__ __align__(16) float Uarr[UV_ROWS * UV_STRIDE];  // W_a@feats[m] + coord + b0
    __shared__ __align__(16) float Varr[UV_ROWS * UV_STRIDE];  // W_b@feats[l] + coord
    __shared__ float Sacc[HID];
    __shared__ float ylds[HID];

    const int b    = blockIdx.x;
    const int tid  = threadIdx.x;
    const int w    = tid >> 5;        // wave 0..3
    const int lane = tid & 31;
    const int lrow = lane & 15;       // row-in-tile (A) / column (B,C,D)
    const int half = lane >> 4;

    if (tid < HID) Sacc[tid] = 0.0f;

    // ---- Stage x_img[b] (C=256, L=25) into LDS as bf16 bits, layout xs[l][c] ----
    const float* xb = x_img + (size_t)b * (C_SZ * L_SZ);
    for (int i = tid; i < C_SZ * L_SZ; i += 128) {
        int c = i / L_SZ, l = i % L_SZ;            // global layout: c*25 + l
        xs[l * XS_STRIDE + c] = __builtin_bit_cast(unsigned short, (__bf16)xb[i]);
    }
    // zero pad rows 25..31 so A gathers need no guards
    for (int i = tid; i < (XS_ROWS - L_SZ) * XS_STRIDE; i += 128)
        xs[L_SZ * XS_STRIDE + i] = 0;
    __syncthreads();

    const unsigned int* xs32 = (const unsigned int*)xs;   // 2 bf16 per uint, row stride 132

    // ---- U/V GEMM: (25x256) @ (256x64) with bf16 WMMA, 16 tile-jobs over 4 waves ----
    for (int j = 0; j < 4; ++j) {
        int job   = w * 4 + j;
        int mat   = job >> 3;          // 0 -> U (W_a half), 1 -> V (W_b half)
        int Mtile = (job >> 2) & 1;
        int Ntile = job & 3;
        int l_row = Mtile * 16 + lrow;                     // A row for this lane (pad rows = 0)
        int n     = Ntile * 16 + lrow;                     // B column for this lane
        int rbase = l_row * (XS_STRIDE / 2);
        const float* gw = g0_w + (size_t)n * (2 * D_SZ) + (mat ? D_SZ : 0);
        v8f acc = {};
        #pragma unroll
        for (int ks = 0; ks < 8; ++ks) {
            AVec a;
            v16bf bm;
            #pragma unroll
            for (int v = 0; v < 8; ++v) {
                int ka = ks * 32 + a_kidx(v, half);        // even
                a.u[v] = xs32[rbase + (ka >> 1)];          // one ds_load_b32: 2 bf16
                int kb = ks * 32 + a_kidx(v, half);        // B K order matches A
                bm[2*v]   = (__bf16)gw[kb];
                bm[2*v+1] = (__bf16)gw[kb + 1];
            }
            acc = __builtin_amdgcn_wmma_f32_16x16x32_bf16(false, a.v, false, bm,
                                                          (short)0, acc, false, false);
        }
        float* dst = mat ? Varr : Uarr;
        #pragma unroll
        for (int v = 0; v < 8; ++v) {
            int lr = Mtile * 16 + v + 8 * half;            // C/D layout row (<32, unconditional)
            dst[lr * UV_STRIDE + n] = acc[v];
        }
    }
    __syncthreads();

    // ---- coordinate-column (c = 256) rank-1 fixup + fold g0 bias into U ----
    for (int i = tid; i < L_SZ * HID; i += 128) {
        int l = i >> 6, jj = i & 63;
        Uarr[l * UV_STRIDE + jj] += g0_w[jj * (2 * D_SZ) + 256] * (float)l + g0_b[jj];
        Varr[l * UV_STRIDE + jj] += g0_w[jj * (2 * D_SZ) + 513] * (float)l;
    }
    __syncthreads();

    // ---- pre-pack g1 weights (B matrix, 64x64 -> 4 Ntiles x 2 ksteps) in registers ----
    v16bf Bg1[4][2];
    float bb1[4];
    #pragma unroll
    for (int nt = 0; nt < 4; ++nt) {
        int n = nt * 16 + lrow;
        bb1[nt] = g1_b[n];
        #pragma unroll
        for (int ks = 0; ks < 2; ++ks) {
            #pragma unroll
            for (int v = 0; v < 8; ++v) {
                int k = ks * 32 + a_kidx(v, half);
                Bg1[nt][ks][2*v]   = (__bf16)g1_w[n * HID + k];
                Bg1[nt][ks][2*v+1] = (__bf16)g1_w[n * HID + k + 1];
            }
        }
    }

    const float2* U2 = (const float2*)Uarr;   // row stride 33 float2
    const float2* V2 = (const float2*)Varr;

    // ---- main loop: 625 pairs in 40 row-tiles of 16, round-robin over 4 waves ----
    float sacc[4] = {0.0f, 0.0f, 0.0f, 0.0f};
    for (int t = w; t < 40; t += 4) {
        int p = t * 16 + lrow;
        if (p > 624) p = 624;                 // clamp pad rows (masked in reduction)
        int lp = p / L_SZ, mp = p % L_SZ;     // pair = (l, m): h = U[m] + V[l]
        int ub = mp * (UV_STRIDE / 2), vb = lp * (UV_STRIDE / 2);
        v16bf a0, a1;
        #pragma unroll
        for (int v = 0; v < 8; ++v) {
            int kh = a_kidx(v, half) >> 1;    // float2 index within row
            float2 uu = U2[ub + kh], vv = V2[vb + kh];
            a0[2*v]   = (__bf16)leaky(uu.x + vv.x);
            a0[2*v+1] = (__bf16)leaky(uu.y + vv.y);
            uu = U2[ub + 16 + kh]; vv = V2[vb + 16 + kh];
            a1[2*v]   = (__bf16)leaky(uu.x + vv.x);
            a1[2*v+1] = (__bf16)leaky(uu.y + vv.y);
        }
        #pragma unroll
        for (int nt = 0; nt < 4; ++nt) {
            v8f d = {};
            d = __builtin_amdgcn_wmma_f32_16x16x32_bf16(false, a0, false, Bg1[nt][0],
                                                        (short)0, d, false, false);
            d = __builtin_amdgcn_wmma_f32_16x16x32_bf16(false, a1, false, Bg1[nt][1],
                                                        (short)0, d, false, false);
            #pragma unroll
            for (int v = 0; v < 8; ++v) {
                int r = v + 8 * half;                       // D layout row within tile
                float val = leaky(d[v] + bb1[nt]);
                sacc[nt] += (t * 16 + r < 625) ? val : 0.0f; // data select, no branch
            }
        }
    }
    #pragma unroll
    for (int nt = 0; nt < 4; ++nt)
        atomicAdd(&Sacc[nt * 16 + lrow], sacc[nt]);         // ds_add_f32 cross-wave reduce
    __syncthreads();

    // ---- tail layers: post (64x64) then out (32x64), tiny VALU matvecs ----
    if (tid < HID) {
        float acc = post_b[tid];
        #pragma unroll 8
        for (int k = 0; k < HID; ++k) acc += post_w[tid * HID + k] * Sacc[k];
        ylds[tid] = leaky(acc);
    }
    __syncthreads();
    if (tid < OUTU) {
        float acc = out_b[tid];
        #pragma unroll 8
        for (int k = 0; k < HID; ++k) acc += out_w[tid * HID + k] * ylds[k];
        out[b * OUTU + tid] = leaky(acc);
    }
}

extern "C" void kernel_launch(void* const* d_in, const int* in_sizes, int n_in,
                              void* d_out, int out_size, void* d_ws, size_t ws_size,
                              hipStream_t stream) {
    const float* x_img  = (const float*)d_in[0];
    const float* g0_w   = (const float*)d_in[1];
    const float* g0_b   = (const float*)d_in[2];
    const float* g1_w   = (const float*)d_in[3];
    const float* g1_b   = (const float*)d_in[4];
    const float* post_w = (const float*)d_in[5];
    const float* post_b = (const float*)d_in[6];
    const float* out_w  = (const float*)d_in[7];
    const float* out_b  = (const float*)d_in[8];
    float* out = (float*)d_out;
    hipLaunchKernelGGL(relnet_kernel, dim3(B_SZ), dim3(128), 0, stream,
                       x_img, g0_w, g0_b, g1_w, g1_b, post_w, post_b, out_w, out_b, out);
}
